// SelfAttention_v2_4741643894815
// MI455X (gfx1250) — matrix operand order repaired and checked by reading, not compiled, not run on hardware
//
#include <hip/hip_runtime.h>
#include <hip/hip_bf16.h>

// ---------------------------------------------------------------------------
// Self-attention forward on MI455X (gfx1250), bf16 WMMA pipeline.
//   q = x@Wq^T ; k = x@Wk^T ; v = x@Wv^T
//   S = q@k^T ; P = softmax(S/sqrt(D)) ; out = P@v
// All GEMMs are NT (C = A * B^T, K contiguous in both operands) mapped onto
// v_wmma_f32_16x16x32_bf16 with LDS-staged, double-buffered block tiles.
// Next-tile fill uses the gfx1250 async global->LDS engine when the builtins
// are available (ASYNCcnt-tracked), else falls back to register prefetch.
// ---------------------------------------------------------------------------

typedef __attribute__((ext_vector_type(16))) __bf16 v16bf;
typedef __attribute__((ext_vector_type(8)))  __bf16 v8bf;
typedef __attribute__((ext_vector_type(8)))  float  v8f;

union ABFrag {
    v16bf v;
    v8bf  h[2];
};

__device__ __forceinline__ v8bf ld8(const __bf16* p) {
    return *reinterpret_cast<const v8bf*>(p);
}
__device__ __forceinline__ void st8(__bf16* p, v8bf v) {
    *reinterpret_cast<v8bf*>(p) = v;
}

// -------- gfx1250 async global->LDS copy (guarded; fallback = reg prefetch)
#if defined(__has_builtin)
#  if __has_builtin(__builtin_amdgcn_global_load_async_to_lds_b128) && \
      __has_builtin(__builtin_amdgcn_s_wait_asynccnt)
#    define USE_ASYNC_LDS 1
#  endif
#endif
#ifndef USE_ASYNC_LDS
#  define USE_ASYNC_LDS 0
#endif

#if USE_ASYNC_LDS
#define AS1 __attribute__((address_space(1)))
#define AS3 __attribute__((address_space(3)))
typedef int b128_t __attribute__((__vector_size__(4 * sizeof(int))));
__device__ __forceinline__ void async_cp16(const __bf16* g, __bf16* l) {
    // global_load_async_to_lds_b128 (ASYNCcnt-tracked)
    __builtin_amdgcn_global_load_async_to_lds_b128(
        (AS1 b128_t*)(g), (AS3 b128_t*)(l), /*offset=*/0, /*cpol=*/0);
}
#endif

// ---------------------------------------------------------------------------
// fp32 -> bf16 conversion
// ---------------------------------------------------------------------------
__global__ __launch_bounds__(256) void f32_to_bf16_kernel(
    const float* __restrict__ in, __bf16* __restrict__ out, int n)
{
    int i = blockIdx.x * 256 + threadIdx.x;
    if (i < n) out[i] = (__bf16)in[i];
}

// ---------------------------------------------------------------------------
// WMMA NT GEMM: C[M,N] = scale * (A[M,K] * B[N,K]^T)
//   A row-major (lda = K stride), B row-major (ldb = K stride).
// STORE_MODE: 0 = f32 C[m*ldc+n], 1 = bf16 C[m*ldc+n], 2 = bf16 C[n*ldc+m]
//
// Block: 256 threads (8 wave32). Block tile 128(M) x 256(N), K-chunk 32.
// Wave grid 2x4, wave tile 64x64 = 4x4 WMMA tiles (128 accumulator VGPRs).
// A/B tiles staged in LDS (row stride padded to 40 elems = 20 banks ->
// conflict-free 16B fragment reads), double buffered.
// Requires: M % 128 == 0, N % 256 == 0, K % 32 == 0 (true for all calls).
// ---------------------------------------------------------------------------
template <int STORE_MODE>
__global__ __launch_bounds__(256) void wmma_gemm_nt_kernel(
    const __bf16* __restrict__ A, int lda,
    const __bf16* __restrict__ B, int ldb,
    void* __restrict__ Cout, int ldc,
    int K, float scale)
{
    constexpr int BM = 128, BN = 256, BK = 32;
    constexpr int SK = 40;                      // padded LDS row stride (elems)
    __shared__ __align__(16) __bf16 sm[2][(BM + BN) * SK];   // 60 KB

    const int tid   = threadIdx.x;
    const int lane  = tid & 31;                 // wave32
    const int wave  = tid >> 5;                 // 0..7
    const int lrow  = lane & 15;                // M (A) / N (B) index in tile
    const int khalf = lane >> 4;                // which K half this lane holds

    const int wm  = wave & 1;                   // wave grid 2 (M) x 4 (N)
    const int wn  = wave >> 1;
    const int m0l = wm * 64;                    // wave offsets in block tile
    const int n0l = wn * 64;
    const int bM0 = blockIdx.y * BM;
    const int bN0 = blockIdx.x * BN;

    // Cooperative fill mapping: thread t moves b128 chunks at row r0(+64s),
    // K-offset cc. A tile: 2 chunks/thread, B tile: 4 chunks/thread.
    const int cc = (tid & 3) * 8;               // 0,8,16,24 (elements)
    const int r0 = tid >> 2;                    // 0..63
    const __bf16* gA = A + (size_t)(bM0 + r0) * lda + cc;
    const __bf16* gB = B + (size_t)(bN0 + r0) * ldb + cc;
    const int laBase = r0 * SK + cc;            // A region offset in LDS
    const int lbBase = BM * SK + r0 * SK + cc;  // B region offset in LDS

    v8f acc[4][4];
    const v8f vzero = {0.f, 0.f, 0.f, 0.f, 0.f, 0.f, 0.f, 0.f};
#pragma unroll
    for (int i = 0; i < 4; ++i)
#pragma unroll
        for (int j = 0; j < 4; ++j)
            acc[i][j] = vzero;

    const int NT = K / BK;

    // Prologue: fill buffer 0 with K-tile 0
#if USE_ASYNC_LDS
#pragma unroll
    for (int s = 0; s < 2; ++s)
        async_cp16(gA + (size_t)s * 64 * lda, &sm[0][laBase + s * 64 * SK]);
#pragma unroll
    for (int s = 0; s < 4; ++s)
        async_cp16(gB + (size_t)s * 64 * ldb, &sm[0][lbBase + s * 64 * SK]);
    __builtin_amdgcn_s_wait_asynccnt(0);
#else
#pragma unroll
    for (int s = 0; s < 2; ++s)
        st8(&sm[0][laBase + s * 64 * SK], ld8(gA + (size_t)s * 64 * lda));
#pragma unroll
    for (int s = 0; s < 4; ++s)
        st8(&sm[0][lbBase + s * 64 * SK], ld8(gB + (size_t)s * 64 * ldb));
#endif
    __syncthreads();

    for (int kt = 0; kt < NT; ++kt) {
        const int cur   = kt & 1;
        const int nxt   = cur ^ 1;
        const bool more = (kt + 1 < NT);
        const int k0n   = (kt + 1) * BK;

#if USE_ASYNC_LDS
        // Fire-and-forget fill of the other buffer; drains behind the WMMAs.
        if (more) {
#pragma unroll
            for (int s = 0; s < 2; ++s)
                async_cp16(gA + (size_t)s * 64 * lda + k0n,
                           &sm[nxt][laBase + s * 64 * SK]);
#pragma unroll
            for (int s = 0; s < 4; ++s)
                async_cp16(gB + (size_t)s * 64 * ldb + k0n,
                           &sm[nxt][lbBase + s * 64 * SK]);
        }
#else
        // Register prefetch of next K-tile (overlaps with WMMA below)
        v8bf pfA[2], pfB[4];
        if (more) {
#pragma unroll
            for (int s = 0; s < 2; ++s)
                pfA[s] = ld8(gA + (size_t)s * 64 * lda + k0n);
#pragma unroll
            for (int s = 0; s < 4; ++s)
                pfB[s] = ld8(gB + (size_t)s * 64 * ldb + k0n);
        }
#endif
        // Warm L2 one further tile ahead (global_prefetch_b8)
        if (kt + 2 < NT) {
            __builtin_prefetch(gA + (size_t)(kt + 2) * BK, 0, 1);
            __builtin_prefetch(gB + (size_t)(kt + 2) * BK, 0, 1);
        }

        // A fragments from LDS (ISA 7.12.2 16-bit A 16x32 layout):
        // lane L: row M=L%16, K = {base..base+7, base+16..base+23}, base=8*(L/16)
        ABFrag a[4];
        const int colA = khalf * 8;
#pragma unroll
        for (int i = 0; i < 4; ++i) {
            const __bf16* ap = &sm[cur][(m0l + i * 16 + lrow) * SK + colA];
            a[i].h[0] = ld8(ap);
            a[i].h[1] = ld8(ap + 16);
        }

        // B fragments (32x16): lane L: col N=L%16, K = {base..base+7, base+16..base+23}, base=8*(L/16)
        const int colB = khalf * 8;
#pragma unroll
        for (int j = 0; j < 4; ++j) {
            ABFrag b;
            const __bf16* bp =
                &sm[cur][BM * SK + (n0l + j * 16 + lrow) * SK + colB];
            b.h[0] = ld8(bp);
            b.h[1] = ld8(bp + 16);
#pragma unroll
            for (int i = 0; i < 4; ++i)
                acc[i][j] = __builtin_amdgcn_wmma_f32_16x16x32_bf16(
                    /*neg_a=*/false, a[i].v,
                    /*neg_b=*/false, b.v,
                    /*c_mod=*/(short)0, acc[i][j],
                    /*reuse_a=*/false, /*reuse_b=*/false);
        }

#if USE_ASYNC_LDS
        if (more) __builtin_amdgcn_s_wait_asynccnt(0);
#else
        if (more) {
#pragma unroll
            for (int s = 0; s < 2; ++s)
                st8(&sm[nxt][laBase + s * 64 * SK], pfA[s]);
#pragma unroll
            for (int s = 0; s < 4; ++s)
                st8(&sm[nxt][lbBase + s * 64 * SK], pfB[s]);
        }
#endif
        __syncthreads();
    }

    // Epilogue: C/D layout: VGPR r of lane L holds M = r + 8*(L/16), N = L%16
#pragma unroll
    for (int i = 0; i < 4; ++i) {
        const int mbase = bM0 + m0l + i * 16 + khalf * 8;
#pragma unroll
        for (int j = 0; j < 4; ++j) {
            const int n = bN0 + n0l + j * 16 + lrow;
#pragma unroll
            for (int r = 0; r < 8; ++r) {
                const int m = mbase + r;
                const float val = acc[i][j][r] * scale;
                if (STORE_MODE == 0) {
                    reinterpret_cast<float*>(Cout)[(size_t)m * ldc + n] = val;
                } else if (STORE_MODE == 1) {
                    reinterpret_cast<__bf16*>(Cout)[(size_t)m * ldc + n] = (__bf16)val;
                } else {
                    reinterpret_cast<__bf16*>(Cout)[(size_t)n * ldc + m] = (__bf16)val;
                }
            }
        }
    }
}

// ---------------------------------------------------------------------------
// Row softmax: P[row,:] = softmax(S[row,:] * scale), fp32 in, bf16 out.
// ---------------------------------------------------------------------------
__global__ __launch_bounds__(256) void softmax_rows_kernel(
    const float* __restrict__ S, __bf16* __restrict__ P, int ncols, float scale)
{
    __shared__ float red[256];
    const int row = blockIdx.x;
    const int tid = threadIdx.x;
    const float* s = S + (size_t)row * ncols;
    __bf16* p = P + (size_t)row * ncols;

    float mx = -3.0e38f;
    for (int c = tid; c < ncols; c += 256) mx = fmaxf(mx, s[c] * scale);
    red[tid] = mx;
    __syncthreads();
    for (int off = 128; off > 0; off >>= 1) {
        if (tid < off) red[tid] = fmaxf(red[tid], red[tid + off]);
        __syncthreads();
    }
    mx = red[0];
    __syncthreads();

    float sum = 0.f;
    for (int c = tid; c < ncols; c += 256) sum += __expf(s[c] * scale - mx);
    red[tid] = sum;
    __syncthreads();
    for (int off = 128; off > 0; off >>= 1) {
        if (tid < off) red[tid] += red[tid + off];
        __syncthreads();
    }
    const float inv = 1.0f / red[0];

    for (int c = tid; c < ncols; c += 256)
        p[c] = (__bf16)(__expf(s[c] * scale - mx) * inv);
}

// ---------------------------------------------------------------------------
// Host-side orchestration
// ---------------------------------------------------------------------------
extern "C" void kernel_launch(void* const* d_in, const int* in_sizes, int n_in,
                              void* d_out, int out_size, void* d_ws, size_t ws_size,
                              hipStream_t stream) {
    (void)in_sizes; (void)n_in; (void)out_size; (void)ws_size;

    const int N = 4096, D = 1024;
    const float inv_sqrt_d = 0.03125f;   // 1/sqrt(1024)

    const float* x  = (const float*)d_in[0];
    const float* Wq = (const float*)d_in[1];
    const float* Wk = (const float*)d_in[2];
    const float* Wv = (const float*)d_in[3];
    float* out = (float*)d_out;

    // Workspace carve-up (256B aligned slices)
    char* ws = (char*)d_ws;
    size_t off = 0;
    auto carve = [&](size_t bytes) -> void* {
        void* p = ws + off;
        off += (bytes + 255) & ~(size_t)255;
        return p;
    };
    __bf16* xb  = (__bf16*)carve((size_t)N * D * 2);
    __bf16* wqb = (__bf16*)carve((size_t)D * D * 2);
    __bf16* wkb = (__bf16*)carve((size_t)D * D * 2);
    __bf16* wvb = (__bf16*)carve((size_t)D * D * 2);
    __bf16* Qb  = (__bf16*)carve((size_t)N * D * 2);
    __bf16* Kb  = (__bf16*)carve((size_t)N * D * 2);
    __bf16* Vtb = (__bf16*)carve((size_t)D * N * 2);  // V transposed: [D, N]
    float*  Sf  = (float*)carve((size_t)N * N * 4);
    __bf16* Pb  = (__bf16*)carve((size_t)N * N * 2);

    // 1) fp32 -> bf16 conversions
    f32_to_bf16_kernel<<<(N * D + 255) / 256, 256, 0, stream>>>(x, xb, N * D);
    f32_to_bf16_kernel<<<(D * D + 255) / 256, 256, 0, stream>>>(Wq, wqb, D * D);
    f32_to_bf16_kernel<<<(D * D + 255) / 256, 256, 0, stream>>>(Wk, wkb, D * D);
    f32_to_bf16_kernel<<<(D * D + 255) / 256, 256, 0, stream>>>(Wv, wvb, D * D);

    // 2) Projections: Q = x Wq^T, K = x Wk^T (bf16), V^T stored as [D, N]
    dim3 blk(256);
    dim3 gProj(D / 256, N / 128);
    wmma_gemm_nt_kernel<1><<<gProj, blk, 0, stream>>>(xb, D, wqb, D, Qb, D, D, 1.0f);
    wmma_gemm_nt_kernel<1><<<gProj, blk, 0, stream>>>(xb, D, wkb, D, Kb, D, D, 1.0f);
    wmma_gemm_nt_kernel<2><<<gProj, blk, 0, stream>>>(xb, D, wvb, D, Vtb, N, D, 1.0f);

    // 3) Scores: S = Q K^T (fp32, unscaled; scale applied in softmax)
    dim3 gScores(N / 256, N / 128);
    wmma_gemm_nt_kernel<0><<<gScores, blk, 0, stream>>>(Qb, D, Kb, D, Sf, N, D, 1.0f);

    // 4) Row softmax with 1/sqrt(D) scaling -> bf16 P
    softmax_rows_kernel<<<N, blk, 0, stream>>>(Sf, Pb, N, inv_sqrt_d);

    // 5) Output: out = P V = P (Vt)^T  (fp32)
    dim3 gOut(D / 256, N / 128);
    wmma_gemm_nt_kernel<0><<<gOut, blk, 0, stream>>>(Pb, N, Vtb, N, out, D, N, 1.0f);
}
